// SelfAttention2D_80333068305050
// MI455X (gfx1250) — matrix operand order repaired and checked by reading, not compiled, not run on hardware
//
#include <hip/hip_runtime.h>
#include <hip/hip_bf16.h>
#include <math.h>

// Problem constants (match reference)
#define BATCH 4
#define CH    512      // C
#define KD    512      // key dim
#define VD    512      // value dim
#define NT    4096     // H*W tokens

typedef __bf16 bf16;
typedef __attribute__((ext_vector_type(16))) __bf16 v16bf;
typedef __attribute__((ext_vector_type(8)))  __bf16 v8bf;
typedef __attribute__((ext_vector_type(8)))  float  v8f;
typedef __attribute__((ext_vector_type(4)))  int    v4i;

#define AS1 __attribute__((address_space(1)))
#define AS3 __attribute__((address_space(3)))

// Async global->LDS copy path (gfx1250 GLOBAL_LOAD_ASYNC_TO_LDS_B128).
// Probe-verified on this toolchain: takes (v4i AS1*, v4i AS3*, imm, imm cpol).
#if defined(__has_builtin)
#if __has_builtin(__builtin_amdgcn_global_load_async_to_lds_b128)
#define HAVE_ASYNC_LDS 1
#endif
#endif
#ifndef HAVE_ASYNC_LDS
#define HAVE_ASYNC_LDS 0
#endif

__device__ __forceinline__ void wait_async0() {
#if defined(__has_builtin) && __has_builtin(__builtin_amdgcn_s_wait_asynccnt)
  __builtin_amdgcn_s_wait_asynccnt(0);
#else
  asm volatile("s_wait_asynccnt 0x0" ::: "memory");
#endif
}

// ---------------------------------------------------------------------------
// WMMA wrapper: D = A(16x32 bf16) * B(32x16 bf16) + C(16x16 f32)
// ---------------------------------------------------------------------------
__device__ __forceinline__ v8f wmma_bf16(v16bf a, v16bf b, v8f c) {
  return __builtin_amdgcn_wmma_f32_16x16x32_bf16(
      /*neg_a=*/false, a, /*neg_b=*/false, b,
      /*c_mod=*/(short)0, c, /*reuse_a=*/false, /*reuse_b=*/false);
}

// ---------------------------------------------------------------------------
// Fragment loaders (contiguous k = 16h + s order, wave32)
// A (16x32): lane L -> row r=L&15, half h=L>>4; elems = A[r][k0+16h..k0+16h+15]
// B (32x16): lane L -> column n=L&15, half h=L>>4; elems = 16 K-contiguous values
// ---------------------------------------------------------------------------
__device__ __forceinline__ v16bf load_afrag(const bf16* row, int h) {
  v8bf lo = *(const v8bf*)(row + 16 * h);
  v8bf hi = *(const v8bf*)(row + 16 * h + 8);
  v16bf r;
#pragma unroll
  for (int i = 0; i < 8; ++i) { r[i] = lo[i]; r[i + 8] = hi[i]; }
  return r;
}

__device__ __forceinline__ v16bf load_bfrag(const bf16* p) {
  v8bf lo = *(const v8bf*)(p);
  v8bf hi = *(const v8bf*)(p + 8);
  v16bf r;
#pragma unroll
  for (int i = 0; i < 8; ++i) { r[i] = lo[i]; r[i + 8] = hi[i]; }
  return r;
}

// ---------------------------------------------------------------------------
// XOR-lane reductions within 16-lane half-groups (C/D rows are per-VGPR,
// columns per-lane; lanes 0-15 / 16-31 hold independent row halves).
// ds_swizzle group-of-32 imm = (xor<<10)|(or<<5)|and, and=0x1f.
// ---------------------------------------------------------------------------
template <int M>
__device__ __forceinline__ float swz_xor(float x) {
  return __int_as_float(
      __builtin_amdgcn_ds_swizzle(__float_as_int(x), (M << 10) | 0x1f));
}
__device__ __forceinline__ float half_max(float x) {
  x = fmaxf(x, swz_xor<1>(x));
  x = fmaxf(x, swz_xor<2>(x));
  x = fmaxf(x, swz_xor<4>(x));
  x = fmaxf(x, swz_xor<8>(x));
  return x;
}
__device__ __forceinline__ float half_sum(float x) {
  x += swz_xor<1>(x);
  x += swz_xor<2>(x);
  x += swz_xor<4>(x);
  x += swz_xor<8>(x);
  return x;
}

// ---------------------------------------------------------------------------
// Kernel 1: NCHW f32 -> token-major (B*N, C) bf16, LDS-tiled 32x32 transpose
// ---------------------------------------------------------------------------
__global__ __launch_bounds__(256) void cvt_x(const float* __restrict__ x,
                                             bf16* __restrict__ xb) {
  __shared__ float t[32][33];
  const int tid = threadIdx.x;
  const int n0 = blockIdx.x * 32, c0 = blockIdx.y * 32, b = blockIdx.z;
  const int col = tid & 31, rw = tid >> 5;
#pragma unroll
  for (int u = 0; u < 4; ++u) {
    const int cr = rw + u * 8;
    t[cr][col] = x[((size_t)b * CH + c0 + cr) * NT + n0 + col];
  }
  __syncthreads();
#pragma unroll
  for (int u = 0; u < 4; ++u) {
    const int nr = rw + u * 8;
    xb[((size_t)b * NT + n0 + nr) * CH + c0 + col] = (bf16)t[col][nr];
  }
}

// Kernel 2: f32 -> bf16 copy (weights)
__global__ __launch_bounds__(256) void cvt_f(const float* __restrict__ s,
                                             bf16* __restrict__ d, int n) {
  int i = blockIdx.x * 256 + threadIdx.x;
  if (i < n) d[i] = (bf16)s[i];
}

// ---------------------------------------------------------------------------
// Kernel 3: projection GEMM  O[t][d] = sum_c X[t][c] * W[d][c]
// ---------------------------------------------------------------------------
__global__ __launch_bounds__(128) void proj_gemm(const bf16* __restrict__ X,
                                                 const bf16* __restrict__ W,
                                                 bf16* __restrict__ O,
                                                 int transpose_out) {
  const int lane = threadIdx.x & 31, wave = threadIdx.x >> 5;
  const int t0 = blockIdx.x * 64 + wave * 16;
  const int d0 = blockIdx.y * 64;
  const int r = lane & 15, h = lane >> 4;

  const bf16* arow = X + (size_t)(t0 + r) * CH;          // + kc*32 (+16h inside)
  const bf16* wb   = W + (size_t)(d0 + r) * CH + 16 * h; // + f*16*CH + kc*32

  v8f acc[4];
  v8f z = {};
#pragma unroll
  for (int f = 0; f < 4; ++f) acc[f] = z;

  v16bf a0 = load_afrag(arow, h);
  v16bf b0 = load_bfrag(wb);
#pragma unroll
  for (int kc = 0; kc < 16; ++kc) {
    v16bf a1 = (kc < 15) ? load_afrag(arow + (kc + 1) * 32, h) : a0;
#pragma unroll
    for (int f = 0; f < 4; ++f) {
      v16bf b1;
      if (f < 3)        b1 = load_bfrag(wb + (size_t)(f + 1) * 16 * CH + kc * 32);
      else if (kc < 15) b1 = load_bfrag(wb + (kc + 1) * 32);
      else              b1 = b0;
      acc[f] = wmma_bf16(a0, b0, acc[f]);
      b0 = b1;
    }
    a0 = a1;
  }

  if (!transpose_out) {
#pragma unroll
    for (int f = 0; f < 4; ++f)
#pragma unroll
      for (int v = 0; v < 8; ++v)
        O[(size_t)(t0 + v + 8 * h) * KD + d0 + f * 16 + r] = (bf16)acc[f][v];
  } else {
    const int b = t0 / NT, n = t0 % NT;
#pragma unroll
    for (int f = 0; f < 4; ++f)
#pragma unroll
      for (int v = 0; v < 8; ++v)
        O[((size_t)b * KD + d0 + f * 16 + r) * NT + n + v + 8 * h] =
            (bf16)acc[f][v];
  }
}

// ---------------------------------------------------------------------------
// Kernel 4: fused flash attention, balanced wave pairs.
// 256 thr = 8 waves; wave (s = w>>1, dh = w&1): query strip s (16 rows),
// dim half dh (256 of 512 value dims).  Per 32-key block:
//  - all waves async-stage the K tile (32x512 bf16 = 32 KB) into LDS
//  - EACH wave computes the 16-key S fragment kb=dh (16 WMMAs from LDS)
//  - online-softmax stats merged across the pair via tiny LDS exchanges
//    (partial row-max, partial row-sum); both waves then hold identical
//    (m,l) trajectories -> local O rescale, local final normalization
//  - each wave accumulates O += P*V for its 256 dims (16 WMMAs, pipelined
//    V^T loads).  32 WMMAs per wave per block, no idle waves.
// 3 barriers/iter; cross-iteration LDS hazards are covered (Kl written only
// before barrier-1, read only between barrier-1 and barrier-2; Pl written
// only after barrier-2, read only before the next barrier-1).
// ---------------------------------------------------------------------------
__global__ __launch_bounds__(256) void attn(const bf16* __restrict__ Q,
                                            const bf16* __restrict__ K,
                                            const bf16* __restrict__ Vt,
                                            float* __restrict__ out,
                                            float* __restrict__ rowM,
                                            float* __restrict__ rowL) {
  __shared__ __align__(128) bf16 Kl[32][CH];     // 32 KB staged K block
  __shared__ __align__(128) bf16 Pl[4][16][32];  // 4 KB P tiles
  __shared__ float Mx[4][2][16];                 // partial row maxima
  __shared__ float Rs[4][2][16];                 // partial row sums

  const int tid = threadIdx.x;
  const int lane = tid & 31, wave = tid >> 5;
  const int s = wave >> 1, dh = wave & 1;
  const int b = blockIdx.x >> 6, qt = blockIdx.x & 63;
  const int q0 = qt * 64 + s * 16;
  const int r = lane & 15, h = lane >> 4;
  const float is = 0.04419417382415922f;  // 1/sqrt(512)

  // Preload Q A-fragments for this strip (16 x 512 bf16 in VGPRs).
  v16bf qf[16];
  const bf16* qrow = Q + ((size_t)b * NT + q0 + r) * KD;
#pragma unroll
  for (int kc = 0; kc < 16; ++kc) qf[kc] = load_afrag(qrow + kc * 32, h);

  float mR[8], lR[8];
#pragma unroll
  for (int v = 0; v < 8; ++v) { mR[v] = -1e30f; lR[v] = 0.f; }
  v8f o[16];
  v8f z = {};
#pragma unroll
  for (int f = 0; f < 16; ++f) o[f] = z;

  for (int j = 0; j < NT / 32; ++j) {
    const int k0 = j * 32;
    const bf16* Kg = K + ((size_t)b * NT + k0) * KD;  // 32x512 tile

    // ---- cooperative K tile staging: 256 thr x 8 x 16B, coalesced ----
#if HAVE_ASYNC_LDS
#pragma unroll
    for (int u = 0; u < 8; ++u) {
      const int off = (u * 256 + tid) * 8;  // bf16 elements (16B steps)
      __builtin_amdgcn_global_load_async_to_lds_b128(
          (AS1 v4i*)(Kg + off), (AS3 v4i*)(&Kl[0][0] + off), 0, 0);
    }
    wait_async0();
#else
#pragma unroll
    for (int u = 0; u < 8; ++u) {
      const int off = (u * 256 + tid) * 8;
      *(v8bf*)(&Kl[0][0] + off) = *(const v8bf*)(Kg + off);
    }
#endif
    __syncthreads();  // barrier 1: K tile visible

    // ---- S fragment for this wave's 16 keys (kb = dh), from LDS ----
    v8f c = z;
    {
      const bf16* krow = &Kl[dh * 16 + r][16 * h];
      v16bf b0 = load_bfrag(krow);
#pragma unroll
      for (int kc = 0; kc < 16; ++kc) {
        v16bf b1 = (kc < 15) ? load_bfrag(krow + (kc + 1) * 32) : b0;
        c = wmma_bf16(qf[kc], b0, c);
        b0 = b1;
      }
#pragma unroll
      for (int v = 0; v < 8; ++v) c[v] *= is;
    }

    // ---- partial row maxima -> LDS; merge across the pair ----
#pragma unroll
    for (int v = 0; v < 8; ++v) {
      float pm = half_max(c[v]);
      if (r == 0) Mx[s][dh][v + 8 * h] = pm;
    }
    __syncthreads();  // barrier 2: Mx visible

    float scl[8];
#pragma unroll
    for (int v = 0; v < 8; ++v) {
      float mt = fmaxf(Mx[s][0][v + 8 * h], Mx[s][1][v + 8 * h]);
      float mn = fmaxf(mR[v], mt);
      scl[v] = __expf(mR[v] - mn);
      mR[v] = mn;
    }

    // ---- P = exp(S - m): stage bf16 tile + partial row sums ----
#pragma unroll
    for (int v = 0; v < 8; ++v) {
      float p = __expf(c[v] - mR[v]);
      Pl[s][v + 8 * h][dh * 16 + r] = (bf16)p;
      float ps = half_sum(p);
      if (r == 0) Rs[s][dh][v + 8 * h] = ps;
    }
    __syncthreads();  // barrier 3: P tile + Rs visible

#pragma unroll
    for (int v = 0; v < 8; ++v)
      lR[v] = lR[v] * scl[v] + Rs[s][0][v + 8 * h] + Rs[s][1][v + 8 * h];

    // ---- rescale O (local scl), then O += P * V (pipelined V^T loads) ----
#pragma unroll
    for (int v = 0; v < 8; ++v) {
      float sc = scl[v];
#pragma unroll
      for (int f = 0; f < 16; ++f) o[f][v] *= sc;
    }
    {
      v16bf pa = load_afrag(&Pl[s][r][0], h);
      const bf16* vbase = Vt + ((size_t)b * KD + dh * 256) * NT + k0 + 16 * h;
      v16bf b0 = load_bfrag(vbase + (size_t)r * NT);
#pragma unroll
      for (int f = 0; f < 16; ++f) {
        v16bf b1 =
            (f < 15) ? load_bfrag(vbase + (size_t)((f + 1) * 16 + r) * NT) : b0;
        o[f] = wmma_bf16(pa, b0, o[f]);
        b0 = b1;
      }
    }
    // prefetch next key block (global_prefetch_b8)
    if (j < NT / 32 - 1 && lane < 8)
      __builtin_prefetch(K + ((size_t)b * NT + k0 + 32 + lane * 4) * KD, 0, 1);
    // no end-of-loop barrier needed: next iteration's Kl writes precede
    // barrier 1 and Kl is unread here; Pl writes happen after barrier 2.
  }

  // ---- epilogue: both waves hold identical (m,l); normalize + store ----
  if (dh == 0 && r == 0) {
#pragma unroll
    for (int v = 0; v < 8; ++v) {
      rowM[(size_t)b * NT + q0 + v + 8 * h] = mR[v];
      rowL[(size_t)b * NT + q0 + v + 8 * h] = lR[v];
    }
  }
#pragma unroll
  for (int f = 0; f < 16; ++f) {
    const int d = dh * 256 + f * 16 + r;
#pragma unroll
    for (int v = 0; v < 8; ++v)
      out[((size_t)b * VD + d) * NT + q0 + v + 8 * h] = o[f][v] / lR[v];
  }
}

// ---------------------------------------------------------------------------
// Kernel 5: scores_out[b][m] = sum_q exp(S[q][m] - m_q) / l_q.
// S^T tiles via WMMA (A = K rows, B = Q^T); per-lane columns are queries so
// (m_q, l_q) are per-lane scalars; XOR-sum reduces 16 query lanes per row.
// ---------------------------------------------------------------------------
__global__ __launch_bounds__(128) void score_sums(const bf16* __restrict__ Q,
                                                  const bf16* __restrict__ K,
                                                  const float* __restrict__ rowM,
                                                  const float* __restrict__ rowL,
                                                  float* __restrict__ sout) {
  const int lane = threadIdx.x & 31, wave = threadIdx.x >> 5;
  const int b = blockIdx.x >> 6, kt = blockIdx.x & 63;
  const int kbase = kt * 64 + wave * 16;
  const int r = lane & 15, h = lane >> 4;
  const float is = 0.04419417382415922f;

  v16bf kf[16];
#pragma unroll
  for (int kc = 0; kc < 16; ++kc)
    kf[kc] = load_afrag(K + ((size_t)b * NT + kbase + r) * KD + kc * 32, h);

  float acc[8];
#pragma unroll
  for (int v = 0; v < 8; ++v) acc[v] = 0.f;
  v8f z = {};

  for (int qblk = 0; qblk < NT / 16; ++qblk) {
    const int q0 = qblk * 16;
    const bf16* qbase = Q + ((size_t)b * NT + q0 + r) * KD + 16 * h;
    v8f c = z;
    v16bf b0 = load_bfrag(qbase);
#pragma unroll
    for (int kc = 0; kc < 16; ++kc) {
      v16bf b1 = (kc < 15) ? load_bfrag(qbase + (kc + 1) * 32) : b0;
      c = wmma_bf16(kf[kc], b0, c);
      b0 = b1;
    }
    float mq = rowM[(size_t)b * NT + q0 + r];
    float li = 1.f / rowL[(size_t)b * NT + q0 + r];
#pragma unroll
    for (int v = 0; v < 8; ++v) acc[v] += __expf(c[v] * is - mq) * li;
  }
#pragma unroll
  for (int v = 0; v < 8; ++v) {
    float sv = half_sum(acc[v]);
    if (r == 0) sout[(size_t)b * NT + kbase + v + 8 * h] = sv;
  }
}

// ---------------------------------------------------------------------------
// Host-side launch
// ---------------------------------------------------------------------------
extern "C" void kernel_launch(void* const* d_in, const int* in_sizes, int n_in,
                              void* d_out, int out_size, void* d_ws,
                              size_t ws_size, hipStream_t stream) {
  (void)in_sizes; (void)n_in; (void)out_size; (void)ws_size;
  const float* x  = (const float*)d_in[0];
  const float* Wk = (const float*)d_in[1];
  const float* Wq = (const float*)d_in[2];
  const float* Wv = (const float*)d_in[3];

  char* ws = (char*)d_ws;
  const size_t SZ_X  = (size_t)BATCH * NT * CH * sizeof(bf16);   // 16 MB
  const size_t SZ_W  = (size_t)KD * CH * sizeof(bf16);           // 0.5 MB
  const size_t SZ_QK = (size_t)BATCH * NT * KD * sizeof(bf16);   // 16 MB
  bf16* Xb  = (bf16*)(ws);
  bf16* Wqb = (bf16*)(ws + SZ_X);
  bf16* Wkb = (bf16*)(ws + SZ_X + SZ_W);
  bf16* Wvb = (bf16*)(ws + SZ_X + 2 * SZ_W);
  bf16* Qb  = (bf16*)(ws + SZ_X + 3 * SZ_W);
  bf16* Kb  = (bf16*)(ws + SZ_X + 3 * SZ_W + SZ_QK);
  bf16* Vt  = (bf16*)(ws + SZ_X + 3 * SZ_W + 2 * SZ_QK);
  float* rM = (float*)(ws + SZ_X + 3 * SZ_W + 3 * SZ_QK);
  float* rL = rM + (size_t)BATCH * NT;

  float* out  = (float*)d_out;
  float* sout = out + (size_t)BATCH * VD * NT;  // scores_out after main out

  dim3 cg(NT / 32, CH / 32, BATCH);
  cvt_x<<<cg, 256, 0, stream>>>(x, Xb);
  cvt_f<<<(KD * CH) / 256, 256, 0, stream>>>(Wq, Wqb, KD * CH);
  cvt_f<<<(KD * CH) / 256, 256, 0, stream>>>(Wk, Wkb, KD * CH);
  cvt_f<<<(KD * CH) / 256, 256, 0, stream>>>(Wv, Wvb, KD * CH);

  dim3 pg(BATCH * NT / 64, KD / 64);
  proj_gemm<<<pg, 128, 0, stream>>>(Xb, Wqb, Qb, 0);
  proj_gemm<<<pg, 128, 0, stream>>>(Xb, Wkb, Kb, 0);
  proj_gemm<<<pg, 128, 0, stream>>>(Xb, Wvb, Vt, 1);

  attn<<<BATCH * NT / 64, 256, 0, stream>>>(Qb, Kb, Vt, out, rM, rL);
  score_sums<<<BATCH * NT / 64, 128, 0, stream>>>(Qb, Kb, rM, rL, sout);
}
